// MambaTSADForecast_88519275970829
// MI455X (gfx1250) — matrix operand order repaired and checked by reading, not compiled, not run on hardware
//
#include <hip/hip_runtime.h>
#include <hip/hip_bf16.h>
#include <math.h>

// ---------------------------------------------------------------------------
// Model constants
// ---------------------------------------------------------------------------
#define BATCH     32
#define SEQ_LEN   512
#define INPUT_DIM 64
#define DM        512          // D_MODEL
#define DI        1024         // D_INNER
#define DTR       32           // DT_RANK
#define DS        16           // D_STATE
#define NTOK      64           // tokens per batch (= INPUT_DIM)
#define ROWS      (BATCH*NTOK) // 2048
#define PRED      96

// ---------------------------------------------------------------------------
// bf16 helpers
// ---------------------------------------------------------------------------
typedef __attribute__((ext_vector_type(16))) __bf16 bf16x16;
typedef __attribute__((ext_vector_type(8)))  float  floatx8;

struct alignas(16) U128 { unsigned int x, y, z, w; };
union  FragU { U128 q[2]; bf16x16 v; };

__device__ __forceinline__ unsigned short f2bf(float f) {
  unsigned int u = __builtin_bit_cast(unsigned int, f);
  u += 0x7FFFu + ((u >> 16) & 1u);           // round-to-nearest-even
  return (unsigned short)(u >> 16);
}

template<int ACT>
__device__ __forceinline__ float act_apply(float v) {
  if (ACT == 1) return 0.5f * v * (1.0f + erff(v * 0.70710678118654752f)); // exact GELU
  if (ACT == 2) return (v > 20.0f) ? v : log1pf(expf(v));                  // softplus
  return v;
}

// CDNA5 async copy: global -> LDS, 16B per lane, tracked by ASYNCcnt.
__device__ __forceinline__ void async_b128(unsigned lds_off, const void* gaddr) {
  asm volatile("global_load_async_to_lds_b128 %0, %1, off"
               :: "v"(lds_off), "v"(gaddr) : "memory");
}
__device__ __forceinline__ void wait_async0() {
  asm volatile("s_wait_asynccnt 0" ::: "memory");
}

// ---------------------------------------------------------------------------
// Generic linear-layer GEMM:  C[M,N] = act(A[M,K] . W[N,K]^T + bias)
// A, W are pre-converted bf16. M is always 2048. Block tile 128x64, 8 waves
// (wave32): 4x2 wave grid, each wave a 32x32 tile = 2x2 v_wmma_f32_16x16x32_bf16.
// LDS tiles are double-buffered and filled with global_load_async_to_lds_b128;
// compute on tile k overlaps the async stream of tile k+1 (ASYNCcnt + barrier).
// ---------------------------------------------------------------------------
template<int ACT, bool ACCUM>
__global__ __launch_bounds__(256) void gemm_bf16_wmma(
    const unsigned short* __restrict__ A, int lda,
    const unsigned short* __restrict__ W, int ldw,
    const float* __restrict__ bias,
    float* __restrict__ C, int ldc,            // nullable
    unsigned short* __restrict__ Cbf,          // nullable bf16 mirror of C
    int N, int K)
{
  __shared__ __align__(16) unsigned short As[2][128 * 40]; // 32 cols + pad (80B rows)
  __shared__ __align__(16) unsigned short Bs[2][ 64 * 40];

  const int tid  = threadIdx.x;
  const int lane = tid & 31;
  const int wid  = tid >> 5;
  const int m0   = blockIdx.x * 128;
  const int n0   = blockIdx.y * 64;
  const int wm   = (wid >> 1) * 32;   // 4 waves along M
  const int wn   = (wid & 1)  * 32;   // 2 waves along N
  const int half = lane >> 4;
  const int l15  = lane & 15;

  const unsigned aBase = (unsigned)(unsigned long long)(&As[0][0]);
  const unsigned bBase = (unsigned)(unsigned long long)(&Bs[0][0]);

  auto issue_tile = [&](int kt, int buf) {
    const int k0 = kt << 5;
    // A tile: 128 rows x 64B = 512 x b128 chunks, 2 per thread
    #pragma unroll
    for (int i = 0; i < 2; ++i) {
      int t = tid + (i << 8);                  // 0..511
      int r = t >> 2, c = t & 3;
      unsigned loff = aBase + (unsigned)(buf * (128 * 40 * 2) + r * 80 + c * 16);
      async_b128(loff, A + (size_t)(m0 + r) * lda + k0 + c * 8);
    }
    // W tile: 64 rows x 64B = 256 x b128 chunks, 1 per thread
    {
      int r = tid >> 2, c = tid & 3;
      int gn = n0 + r;
      if (gn < N) {
        unsigned loff = bBase + (unsigned)(buf * (64 * 40 * 2) + r * 80 + c * 16);
        async_b128(loff, W + (size_t)gn * ldw + k0 + c * 8);
      }
    }
  };

  floatx8 acc[2][2] = {};
  const int KT = K >> 5;

  issue_tile(0, 0);
  for (int kt = 0; kt < KT; ++kt) {
    const int cur = kt & 1;
    wait_async0();          // my async stores for tile kt have landed in LDS
    __syncthreads();        // ... and everyone else's too
    if (kt + 1 < KT) issue_tile(kt + 1, cur ^ 1);  // stream next tile during compute

    // A frag: lane = M row; elems 0..7 -> K = 8*half + e, elems 8..15 -> K = 16 + 8*half + e
    // B frag: lane = N col; elems 0..7 -> K = 8*half + e, elems 8..15 -> K = 16 + 8*half + e (W rows are B^T columns)
    FragU aF[2], bF[2];
    #pragma unroll
    for (int t = 0; t < 2; ++t) {
      const unsigned short* ar = &As[cur][(wm + t * 16 + l15) * 40];
      aF[t].q[0] = *(const U128*)(ar + 8 * half);
      aF[t].q[1] = *(const U128*)(ar + 16 + 8 * half);
      const unsigned short* br = &Bs[cur][(wn + t * 16 + l15) * 40];
      bF[t].q[0] = *(const U128*)(br + 8 * half);
      bF[t].q[1] = *(const U128*)(br + 16 + 8 * half);
    }
    #pragma unroll
    for (int i = 0; i < 2; ++i)
      #pragma unroll
      for (int j = 0; j < 2; ++j)
        acc[i][j] = __builtin_amdgcn_wmma_f32_16x16x32_bf16(
            false, aF[i].v, false, bF[j].v, (short)0, acc[i][j], false, false);
  }

  // C/D layout: lane<16 -> M=r, N=lane ; lane>=16 -> M=8+r, N=lane-16
  #pragma unroll
  for (int i = 0; i < 2; ++i) {
    #pragma unroll
    for (int j = 0; j < 2; ++j) {
      int gn = n0 + wn + j * 16 + l15;
      if (gn >= N) continue;
      float bv = bias ? bias[gn] : 0.0f;
      #pragma unroll
      for (int r = 0; r < 8; ++r) {
        int gm = m0 + wm + i * 16 + half * 8 + r;
        float v = act_apply<ACT>(acc[i][j][r] + bv);
        size_t ci = (size_t)gm * ldc + gn;
        if (C) { if (ACCUM) C[ci] += v; else C[ci] = v; }
        if (Cbf) Cbf[ci] = f2bf(v);
      }
    }
  }
}

// ---------------------------------------------------------------------------
// One-shot fp32 -> bf16 conversion of all GEMM weight matrices (32 segments).
// ---------------------------------------------------------------------------
#define NSEG 32
struct WSeg { const float* src[NSEG]; unsigned int off[NSEG + 1]; };

__global__ __launch_bounds__(256) void convert_weights_kernel(
    WSeg s, unsigned short* __restrict__ dst, unsigned int total)
{
  unsigned int i = blockIdx.x * 256 + threadIdx.x;
  if (i >= total) return;
  int lo = 0, hi = NSEG;
  while (hi - lo > 1) { int mid = (lo + hi) >> 1; if (i >= s.off[mid]) lo = mid; else hi = mid; }
  dst[i] = f2bf(s.src[lo][i - s.off[lo]]);
}

// ---------------------------------------------------------------------------
// Per-(b,channel) normalization over the 512-long sequence; writes the
// transposed bf16 view xnT[(b*64+d), l] consumed by the embedding GEMM.
// ---------------------------------------------------------------------------
__global__ __launch_bounds__(256) void normalize_kernel(
    const float* __restrict__ x, unsigned short* __restrict__ xnT,
    float* __restrict__ meanv, float* __restrict__ stdv, float* __restrict__ xlast)
{
  __shared__ float s1[256], s2[256];
  int row = blockIdx.x;                 // b*64 + d
  int b = row >> 6, d = row & 63;
  const float* xp = x + (size_t)b * SEQ_LEN * INPUT_DIM + d;
  float s = 0.f, sq = 0.f;
  for (int l = threadIdx.x; l < SEQ_LEN; l += 256) {
    float v = xp[(size_t)l * INPUT_DIM];
    s += v; sq += v * v;
  }
  s1[threadIdx.x] = s; s2[threadIdx.x] = sq;
  __syncthreads();
  for (int o = 128; o > 0; o >>= 1) {
    if (threadIdx.x < o) { s1[threadIdx.x] += s1[threadIdx.x + o]; s2[threadIdx.x] += s2[threadIdx.x + o]; }
    __syncthreads();
  }
  float mu   = s1[0] * (1.0f / SEQ_LEN);
  float var  = s2[0] * (1.0f / SEQ_LEN) - mu * mu;
  float rstd = rsqrtf(var + 1e-5f);
  if (threadIdx.x == 0) { meanv[row] = mu; stdv[row] = sqrtf(var + 1e-5f); }
  for (int l = threadIdx.x; l < SEQ_LEN; l += 256) {
    float v = (xp[(size_t)l * INPUT_DIM] - mu) * rstd;
    xnT[(size_t)row * SEQ_LEN + l] = f2bf(v);
    if (l == SEQ_LEN - 1) xlast[row] = v;
  }
}

// LayerNorm over last dim (512); bf16 output (consumed only by GEMMs / flip).
__global__ __launch_bounds__(256) void layernorm_kernel(
    const float* __restrict__ in, const float* __restrict__ g,
    const float* __restrict__ bta, unsigned short* __restrict__ out)
{
  __shared__ float s1[256], s2[256];
  int row = blockIdx.x;
  int t = threadIdx.x;
  const float* ip = in + (size_t)row * DM;
  float v0 = ip[t], v1 = ip[t + 256];
  s1[t] = v0 + v1;
  s2[t] = v0 * v0 + v1 * v1;
  __syncthreads();
  for (int o = 128; o > 0; o >>= 1) {
    if (t < o) { s1[t] += s1[t + o]; s2[t] += s2[t + o]; }
    __syncthreads();
  }
  float mu  = s1[0] * (1.0f / DM);
  float var = s2[0] * (1.0f / DM) - mu * mu;
  float rs  = rsqrtf(var + 1e-5f);
  out[(size_t)row * DM + t]       = f2bf((v0 - mu) * rs * g[t]       + bta[t]);
  out[(size_t)row * DM + t + 256] = f2bf((v1 - mu) * rs * g[t + 256] + bta[t + 256]);
}

// Depthwise causal conv (width 4) over the 64-token axis + SiLU.
// fp32 copy feeds the scan; bf16 copy feeds the x_proj GEMM.
__global__ __launch_bounds__(256) void conv_silu_kernel(
    const float* __restrict__ xz, const float* __restrict__ conv_w,
    const float* __restrict__ conv_b, float* __restrict__ xi_f,
    unsigned short* __restrict__ xi_bf)
{
  int idx = blockIdx.x * 256 + threadIdx.x;   // < 2048*1024
  int c = idx & (DI - 1);
  int row = idx >> 10;
  int b = row >> 6, t = row & 63;
  float acc = conv_b[c];
  #pragma unroll
  for (int k = 0; k < 4; ++k) {
    int tt = t - 3 + k;
    if (tt >= 0) acc += xz[(size_t)((b << 6) + tt) * (2 * DI) + c] * conv_w[c * 4 + k];
  }
  float v = acc / (1.0f + expf(-acc));
  xi_f[idx]  = v;
  xi_bf[idx] = f2bf(v);
}

// Selective scan: one thread per (batch, inner-channel); 16 states in regs.
__global__ __launch_bounds__(256) void scan_kernel(
    const float* __restrict__ delta, const float* __restrict__ xi,
    const float* __restrict__ xdb,  const float* __restrict__ A_log,
    const float* __restrict__ Dp,   float* __restrict__ y)
{
  int id  = blockIdx.x * 256 + threadIdx.x;   // < 32*1024
  int dch = id & (DI - 1);
  int b   = id >> 10;
  float Arow[DS], h[DS];
  #pragma unroll
  for (int s = 0; s < DS; ++s) { Arow[s] = -expf(A_log[dch * DS + s]); h[s] = 0.f; }
  float Dv = Dp[dch];
  for (int t = 0; t < NTOK; ++t) {
    int row = (b << 6) + t;
    float dl = delta[(size_t)row * DI + dch];
    float xv = xi[(size_t)row * DI + dch];
    const float* xr = xdb + (size_t)row * 64;      // [dt(32) | B(16) | C(16)]
    float acc = 0.f;
    #pragma unroll
    for (int s = 0; s < DS; ++s) {
      float dA = expf(dl * Arow[s]);
      h[s] = dA * h[s] + dl * xr[32 + s] * xv;
      acc += h[s] * xr[48 + s];
    }
    y[(size_t)row * DI + dch] = acc + xv * Dv;
  }
}

// out_bf = bf16( y * silu(g) ), g = xz[:, 1024:2048]; feeds out_proj GEMM.
__global__ __launch_bounds__(256) void gate_mul_kernel(
    const float* __restrict__ y, const float* __restrict__ xz,
    unsigned short* __restrict__ out_bf)
{
  int idx = blockIdx.x * 256 + threadIdx.x;   // < 2048*1024
  int c = idx & (DI - 1);
  int row = idx >> 10;
  float g = xz[(size_t)row * (2 * DI) + DI + c];
  out_bf[idx] = f2bf(y[idx] * g / (1.0f + expf(-g)));
}

// Flip token axis (within each batch block of 64 rows) of a 2048x512 bf16 tensor.
__global__ __launch_bounds__(256) void flip_kernel(
    const unsigned short* __restrict__ in, unsigned short* __restrict__ out)
{
  int idx = blockIdx.x * 256 + threadIdx.x;   // < 2048*512
  int c = idx & (DM - 1);
  int row = idx >> 9;
  int b = row >> 6, t = row & 63;
  out[idx] = in[(size_t)((b << 6) + (63 - t)) * DM + c];
}

// e += 0.5*(yF + flip(yB))
__global__ __launch_bounds__(256) void residual_kernel(
    float* __restrict__ e, const float* __restrict__ yf, const float* __restrict__ yb)
{
  int idx = blockIdx.x * 256 + threadIdx.x;   // < 2048*512
  int c = idx & (DM - 1);
  int row = idx >> 9;
  int b = row >> 6, t = row & 63;
  e[idx] += 0.5f * (yf[idx] + yb[(size_t)((b << 6) + (63 - t)) * DM + c]);
}

// out[b,p,d] = (yp[(b*64+d),p] + xn_last) * std + mean
__global__ __launch_bounds__(256) void final_kernel(
    const float* __restrict__ yp, const float* __restrict__ xlast,
    const float* __restrict__ stdv, const float* __restrict__ meanv,
    float* __restrict__ out)
{
  int idx = blockIdx.x * 256 + threadIdx.x;   // < 32*96*64
  if (idx >= BATCH * PRED * INPUT_DIM) return;
  int d = idx & 63;
  int p = (idx >> 6) % PRED;
  int b = idx / (PRED * INPUT_DIM);
  int row = (b << 6) + d;
  out[idx] = (yp[(size_t)row * PRED + p] + xlast[row]) * stdv[row] + meanv[row];
}

// ---------------------------------------------------------------------------
// Host-side orchestration
// ---------------------------------------------------------------------------
struct MambaW {
  const float *in_proj, *conv_w, *conv_b, *x_proj, *dt_w, *dt_b, *A_log, *D, *out_proj;
};
struct MambaWbf {  // bf16 copies of the 4 GEMM weights
  const unsigned short *in_proj, *x_proj, *dt_w, *out_proj;
};

static void run_mamba(const unsigned short* zin_bf, const MambaW& w, const MambaWbf& wb,
                      float* out, float* xz, float* xc, unsigned short* xcb,
                      float* xdb, unsigned short* xdbb, float* delta, float* yscan,
                      unsigned short* ysb, hipStream_t st)
{
  dim3 blk(256);
  // xz = z @ in_proj^T                           M=2048 N=2048 K=512
  gemm_bf16_wmma<0, false><<<dim3(16, 32), blk, 0, st>>>(zin_bf, DM, wb.in_proj, DM, nullptr, xz, 2 * DI, nullptr, 2 * DI, DM);
  conv_silu_kernel<<<8192, blk, 0, st>>>(xz, w.conv_w, w.conv_b, xc, xcb);
  // xdb = silu(conv(xi)) @ x_proj^T              M=2048 N=64  K=1024
  gemm_bf16_wmma<0, false><<<dim3(16, 1), blk, 0, st>>>(xcb, DI, wb.x_proj, DI, nullptr, xdb, 64, xdbb, 64, DI);
  // delta = softplus(xdb[:, :32] @ dt_w^T + b)   M=2048 N=1024 K=32
  gemm_bf16_wmma<2, false><<<dim3(16, 16), blk, 0, st>>>(xdbb, 64, wb.dt_w, DTR, w.dt_b, delta, DI, nullptr, DI, DTR);
  scan_kernel<<<128, blk, 0, st>>>(delta, xc, xdb, w.A_log, w.D, yscan);
  gate_mul_kernel<<<8192, blk, 0, st>>>(yscan, xz, ysb);
  // out = y @ out_proj^T                         M=2048 N=512 K=1024
  gemm_bf16_wmma<0, false><<<dim3(16, 8), blk, 0, st>>>(ysb, DI, wb.out_proj, DI, nullptr, out, DM, nullptr, DM, DI);
}

extern "C" void kernel_launch(void* const* d_in, const int* in_sizes, int n_in,
                              void* d_out, int out_size, void* d_ws, size_t ws_size,
                              hipStream_t stream)
{
  // ---- unpack inputs in setup_inputs() dict insertion order ----
  int ii = 0;
  const float* x     = (const float*)d_in[ii++];
  const float* emb_w = (const float*)d_in[ii++];
  const float* emb_b = (const float*)d_in[ii++];

  struct LayerW {
    const float *vcg, *vcb; MambaW mf, mb;
    const float *tdg, *tdb, *fc1w, *fc1b, *fc2w, *fc2b;
  } L[3];

  auto get_mamba = [&](MambaW& m) {
    m.in_proj  = (const float*)d_in[ii++];
    m.conv_w   = (const float*)d_in[ii++];
    m.conv_b   = (const float*)d_in[ii++];
    m.x_proj   = (const float*)d_in[ii++];
    m.dt_w     = (const float*)d_in[ii++];
    m.dt_b     = (const float*)d_in[ii++];
    m.A_log    = (const float*)d_in[ii++];
    m.D        = (const float*)d_in[ii++];
    m.out_proj = (const float*)d_in[ii++];
  };
  for (int l = 0; l < 3; ++l) {
    L[l].vcg = (const float*)d_in[ii++];
    L[l].vcb = (const float*)d_in[ii++];
    get_mamba(L[l].mf);
    get_mamba(L[l].mb);
    L[l].tdg  = (const float*)d_in[ii++];
    L[l].tdb  = (const float*)d_in[ii++];
    L[l].fc1w = (const float*)d_in[ii++];
    L[l].fc1b = (const float*)d_in[ii++];
    L[l].fc2w = (const float*)d_in[ii++];
    L[l].fc2b = (const float*)d_in[ii++];
  }
  const float* encg  = (const float*)d_in[ii++];
  const float* encb  = (const float*)d_in[ii++];
  const float* projw = (const float*)d_in[ii++];
  const float* projb = (const float*)d_in[ii++];

  // ---- carve workspace ----
  char* wp = (char*)d_ws;
  auto carveF = [&](size_t n) { float* p = (float*)wp; wp += ((n * 4 + 255) / 256) * 256; return p; };
  auto carveH = [&](size_t n) { unsigned short* p = (unsigned short*)wp; wp += ((n * 2 + 255) / 256) * 256; return p; };

  // bf16 weight arena: register the 32 GEMM weight matrices
  WSeg seg;
  unsigned int tot = 0;
  int ns = 0;
  auto add_seg = [&](const float* p, unsigned int n) { seg.src[ns] = p; seg.off[ns] = tot; tot += n; ++ns; };
  add_seg(emb_w, DM * SEQ_LEN);
  for (int l = 0; l < 3; ++l) {
    add_seg(L[l].mf.in_proj,  2 * DI * DM);
    add_seg(L[l].mf.x_proj,   64 * DI);
    add_seg(L[l].mf.dt_w,     DI * DTR);
    add_seg(L[l].mf.out_proj, DM * DI);
    add_seg(L[l].mb.in_proj,  2 * DI * DM);
    add_seg(L[l].mb.x_proj,   64 * DI);
    add_seg(L[l].mb.dt_w,     DI * DTR);
    add_seg(L[l].mb.out_proj, DM * DI);
    add_seg(L[l].fc1w,        4 * DM * DM);
    add_seg(L[l].fc2w,        DM * 4 * DM);
  }
  add_seg(projw, PRED * DM);
  seg.off[ns] = tot;                               // ns == 32
  unsigned short* warena = carveH(tot);
  const unsigned short* wemb = warena + seg.off[0];
  MambaWbf WB[3][2];
  const unsigned short *wfc1[3], *wfc2[3];
  for (int l = 0; l < 3; ++l) {
    int base = 1 + l * 10;
    WB[l][0] = { warena + seg.off[base + 0], warena + seg.off[base + 1],
                 warena + seg.off[base + 2], warena + seg.off[base + 3] };
    WB[l][1] = { warena + seg.off[base + 4], warena + seg.off[base + 5],
                 warena + seg.off[base + 6], warena + seg.off[base + 7] };
    wfc1[l] = warena + seg.off[base + 8];
    wfc2[l] = warena + seg.off[base + 9];
  }
  const unsigned short* wproj = warena + seg.off[31];

  // activations
  unsigned short* xnT  = carveH((size_t)ROWS * SEQ_LEN);
  float* meanv = carveF(ROWS);
  float* stdv  = carveF(ROWS);
  float* xlast = carveF(ROWS);
  float* e     = carveF((size_t)ROWS * DM);
  unsigned short* zbf  = carveH((size_t)ROWS * DM);
  unsigned short* zfbf = carveH((size_t)ROWS * DM);
  float* outF  = carveF((size_t)ROWS * DM);
  float* outB  = carveF((size_t)ROWS * DM);
  float* xz    = carveF((size_t)ROWS * 2 * DI);
  float* xc    = carveF((size_t)ROWS * DI);
  unsigned short* xcb  = carveH((size_t)ROWS * DI);
  float* xdb   = carveF((size_t)ROWS * 64);
  unsigned short* xdbb = carveH((size_t)ROWS * 64);
  float* delta = carveF((size_t)ROWS * DI);
  float* yscan = carveF((size_t)ROWS * DI);
  unsigned short* ysb  = carveH((size_t)ROWS * DI);
  unsigned short* h1bf = carveH((size_t)ROWS * 4 * DM);
  float* yp    = carveF((size_t)ROWS * PRED);

  dim3 blk(256);

  // ---- forward pass ----
  convert_weights_kernel<<<(tot + 255) / 256, blk, 0, stream>>>(seg, warena, tot);
  normalize_kernel<<<ROWS, blk, 0, stream>>>(x, xnT, meanv, stdv, xlast);
  // embedding: e = xnT @ emb_w^T + emb_b         M=2048 N=512 K=512
  gemm_bf16_wmma<0, false><<<dim3(16, 8), blk, 0, stream>>>(xnT, SEQ_LEN, wemb, SEQ_LEN, emb_b, e, DM, nullptr, DM, SEQ_LEN);

  for (int l = 0; l < 3; ++l) {
    layernorm_kernel<<<ROWS, blk, 0, stream>>>(e, L[l].vcg, L[l].vcb, zbf);
    flip_kernel<<<4096, blk, 0, stream>>>(zbf, zfbf);
    run_mamba(zbf,  L[l].mf, WB[l][0], outF, xz, xc, xcb, xdb, xdbb, delta, yscan, ysb, stream);
    run_mamba(zfbf, L[l].mb, WB[l][1], outB, xz, xc, xcb, xdb, xdbb, delta, yscan, ysb, stream);
    residual_kernel<<<4096, blk, 0, stream>>>(e, outF, outB);

    layernorm_kernel<<<ROWS, blk, 0, stream>>>(e, L[l].tdg, L[l].tdb, zbf);
    // h1 = gelu(z @ fc1^T + b1)  (bf16 only)     M=2048 N=2048 K=512
    gemm_bf16_wmma<1, false><<<dim3(16, 32), blk, 0, stream>>>(zbf, DM, wfc1[l], DM, L[l].fc1b, nullptr, 4 * DM, h1bf, 4 * DM, DM);
    // e += h1 @ fc2^T + b2                       M=2048 N=512 K=2048
    gemm_bf16_wmma<0, true><<<dim3(16, 8), blk, 0, stream>>>(h1bf, 4 * DM, wfc2[l], 4 * DM, L[l].fc2b, e, DM, nullptr, DM, 4 * DM);
  }

  layernorm_kernel<<<ROWS, blk, 0, stream>>>(e, encg, encb, zbf);
  // yp = z @ proj_w^T + proj_b                   M=2048 N=96 K=512
  gemm_bf16_wmma<0, false><<<dim3(16, 2), blk, 0, stream>>>(zbf, DM, wproj, DM, projb, yp, PRED, nullptr, PRED, DM);
  final_kernel<<<(BATCH * PRED * INPUT_DIM + 255) / 256, blk, 0, stream>>>(
      yp, xlast, stdv, meanv, (float*)d_out);
}
